// MPSSSM_32796370272981
// MI455X (gfx1250) — compile-verified, matrix operand order checked by reading, not run on hardware
//
#include <hip/hip_runtime.h>
#include <cstdint>
#include <cstddef>

typedef __attribute__((ext_vector_type(16))) _Float16 v16h;
typedef __attribute__((ext_vector_type(8)))  _Float16 v8h;
typedef __attribute__((ext_vector_type(8)))  float    v8f;
typedef __attribute__((ext_vector_type(4)))  unsigned int u32x4;
typedef __attribute__((ext_vector_type(8)))  int          i32x8;
typedef __attribute__((ext_vector_type(4)))  int          i32x4;

#define NB     16
#define LSEQ   96
#define EIN    21
#define DMODEL 128
#define DST    32
#define NBINS  8
#define G3N    4104          /* BINS + DS*DM : only part of g3 output that is used */
#define G3T    257           /* ceil(4104/16) tiles (padded to 4112 cols)          */
#define PH2N   4032
#define PH2T   252

#if defined(__HIP_DEVICE_COMPILE__) && __has_builtin(__builtin_amdgcn_tensor_load_to_lds) && __has_builtin(__builtin_amdgcn_s_wait_tensorcnt)
#define HAVE_TDM 1
#else
#define HAVE_TDM 0
#endif

__device__ __forceinline__ float gelu_f(float x) {
  return 0.5f * x * (1.0f + erff(x * 0.70710678118654752440f));
}

// Load A-fragment (16x32 f16) for V_WMMA_F32_16X16X32_F16 from a 16x128 row-major LDS tile.
// lane l: m=l&15, khi=l>>4 ; elems 0..7 -> K=kc*32+khi*8+j ; elems 8..15 -> +16.
__device__ __forceinline__ void load_afrag(const _Float16* src, int kc, int ln, v16h& a) {
  const int m = ln & 15, khi = ln >> 4;
  const int off0 = m * 128 + kc * 32 + khi * 8;
  v8h lo = *(const v8h*)(src + off0);
  v8h hi = *(const v8h*)(src + off0 + 16);
#pragma unroll
  for (int j = 0; j < 8; ++j) { a[j] = lo[j]; a[8 + j] = hi[j]; }
}

// C = A(16x128) x B(128x16) over 4 K-chunks; B pre-packed in fragment order (16 halfs/lane).
__device__ __forceinline__ v8f wmma_kx128(const v16h a[4], const _Float16* __restrict__ pack,
                                          int tile, int ln) {
  v8f c = {0.f, 0.f, 0.f, 0.f, 0.f, 0.f, 0.f, 0.f};
#pragma unroll
  for (int kc = 0; kc < 4; ++kc) {
    const v16h b = *(const v16h*)(pack + (((size_t)tile * 4 + kc) * 32 + ln) * 16);
    c = __builtin_amdgcn_wmma_f32_16x16x32_f16(false, a[kc], false, b, (short)0, c, false, false);
  }
  return c;
}

// ---------------- Kernel 0: h = LN(x@We + be + (x - causal_depthwise_conv(x))@Wr + br) -----
__global__ __launch_bounds__(128) void mps_prep_h(
    const float* __restrict__ x, const float* __restrict__ We, const float* __restrict__ be,
    const float* __restrict__ Wr, const float* __restrict__ br, const float* __restrict__ lg,
    const float* __restrict__ lb, const float* __restrict__ cw, const float* __restrict__ cb,
    float* __restrict__ hout) {
  const int bid = blockIdx.x;
  const int b = bid / LSEQ, l = bid % LSEQ;
  __shared__ float xr[EIN], rr[EIN], red[DMODEL];
  const int d = threadIdx.x;
  if (d < EIN) {
    const int e = d;
    float xv = x[(b * LSEQ + l) * EIN + e];
    float pr = cb[e];
#pragma unroll
    for (int k = 0; k < 3; ++k) {
      int li = l + k - 2;
      float xs = (li >= 0) ? x[(b * LSEQ + li) * EIN + e] : 0.0f;
      pr += xs * cw[e * 3 + k];
    }
    xr[e] = xv;
    rr[e] = xv - pr;
  }
  __syncthreads();
  float acc = be[d] + br[d];
#pragma unroll
  for (int e = 0; e < EIN; ++e) acc += xr[e] * We[e * DMODEL + d] + rr[e] * Wr[e * DMODEL + d];
  red[d] = acc;
  __syncthreads();
#pragma unroll
  for (int o = 64; o > 0; o >>= 1) { if (d < o) red[d] += red[d + o]; __syncthreads(); }
  const float mu = red[0] * (1.0f / 128.0f);
  __syncthreads();
  const float dv = acc - mu;
  red[d] = dv * dv;
  __syncthreads();
#pragma unroll
  for (int o = 64; o > 0; o >>= 1) { if (d < o) red[d] += red[d + o]; __syncthreads(); }
  const float var = red[0] * (1.0f / 128.0f);
  hout[(b * LSEQ + l) * DMODEL + d] = dv * rsqrtf(var + 1e-5f) * lg[d] + lb[d];
}

// ---------------- Kernel 1: pack weights to f16 WMMA B-fragment order ----------------------
__global__ __launch_bounds__(128) void mps_pack(
    const float* __restrict__ g1w, const float* __restrict__ g2w,
    const float* __restrict__ g3w, const float* __restrict__ ph2w,
    _Float16* __restrict__ wG1, _Float16* __restrict__ wG2,
    _Float16* __restrict__ wG3, _Float16* __restrict__ wPh2) {
  const int bidx = blockIdx.x;
  const float* src; _Float16* dst; int t, ld, N;
  if (bidx < 8)        { src = g1w;  dst = wG1;  t = bidx;       ld = 128;  N = 128;  }
  else if (bidx < 16)  { src = g2w;  dst = wG2;  t = bidx - 8;   ld = 128;  N = 128;  }
  else if (bidx < 16 + G3T) { src = g3w; dst = wG3; t = bidx - 16; ld = 8200; N = G3N; }
  else                 { src = ph2w; dst = wPh2; t = bidx - 16 - G3T; ld = PH2N; N = PH2N; }
  const int tid = threadIdx.x, c = tid >> 5, l = tid & 31;
  const int n = t * 16 + (l & 15);
  const int kb = c * 32 + (l >> 4) * 8;
  _Float16* o = dst + (((size_t)t * 4 + c) * 32 + l) * 16;
#pragma unroll
  for (int j = 0; j < 16; ++j) {
    const int k = (j < 8) ? (kb + j) : (kb + 16 + (j - 8));
    o[j] = (n < N) ? (_Float16)src[(size_t)k * ld + n] : (_Float16)0.0f;
  }
}

// ---------------- per-batch recurrence body (noinline: isolates register pressure) ---------
// Handles mix->delta, qv, expm/phi1 via scaling+squaring, smax power iteration, s_lin,
// encoder mu/logvar MLPs, KL -> new (state, rate). One wave == one batch element.
__device__ __noinline__ void mps_step_batch(
    const float* __restrict__ A_base, const float* __restrict__ dt,
    const float* __restrict__ em1_w, const float* __restrict__ em1_b,
    const float* __restrict__ em2_w, const float* __restrict__ em2_b,
    const float* __restrict__ el1_w, const float* __restrict__ el1_b,
    const float* __restrict__ el2_w, const float* __restrict__ el2_b,
    const float* __restrict__ pp_w, const float* __restrict__ pp_b,
    const float* __restrict__ p_lv, const float* __restrict__ Bm,
    float* binsL, float* rateL, float* stateL, float* hbuf,
    float* Ep, float* eib, float* hidb, int b, int ln) {
  // delta from mix
  float e8[NBINS];
  float mx = -1e30f;
#pragma unroll
  for (int i = 0; i < NBINS; ++i) { e8[i] = binsL[b * NBINS + i]; mx = fmaxf(mx, e8[i]); }
  float s = 0.0f;
#pragma unroll
  for (int i = 0; i < NBINS; ++i) { e8[i] = __expf(e8[i] - mx); s += e8[i]; }
  const float inv = 1.0f / s;
  const float rold = rateL[b];
  const float alpha = 1.0f / (1.0f + __expf(-5.0f * (rold - 0.05f)));
  float ssum = 0.0f;
#pragma unroll
  for (int i = 0; i < NBINS; ++i) {
    float m = 0.5f * (e8[i] * inv + (i == NBINS - 1 ? alpha : 0.0f) + (i == 0 ? (1.0f - alpha) : 0.0f));
    e8[i] = m; ssum += m;
  }
  const float rnorm = 1.0f / fmaxf(ssum, 1e-6f);
  float dlt = 0.0f;
#pragma unroll
  for (int i = 0; i < NBINS; ++i) dlt += e8[i] * rnorm * dt[i];

  // qv = Bm @ h_t  (lane i -> row i)
  float qv = 0.0f;
  const float* bmrow = Bm + (size_t)b * 4096 + ln * DMODEL;
  const float* hrow = &hbuf[b * DMODEL];
#pragma unroll 8
  for (int j = 0; j < DMODEL; ++j) qv += bmrow[j] * hrow[j];

  // M = dlt * A_base ; column ln in regs
  float Mc[32];
#pragma unroll
  for (int i = 0; i < 32; ++i) Mc[i] = dlt * A_base[i * 32 + ln];
  float cs = 0.0f;
#pragma unroll
  for (int i = 0; i < 32; ++i) cs += fabsf(Mc[i]);
#pragma unroll
  for (int o = 16; o > 0; o >>= 1) cs = fmaxf(cs, __shfl_xor(cs, o, 32));  // ||M||_1
  int sq = 0;
  if (cs > 0.25f) { sq = (int)ceilf(log2f(cs * 4.0f)); if (sq > 20) sq = 20; }
  const float scl = exp2f(-(float)sq);
#pragma unroll
  for (int i = 0; i < 32; ++i) Mc[i] *= scl;
#pragma unroll
  for (int i = 0; i < 32; ++i) Ep[i * 32 + ln] = Mc[i];   // Ms shared for lane-invariant reads
  __threadfence_block();

  // Taylor: E = e^Ms, Phi = phi1(Ms), order 6
  float Pc[32], Ec[32], Phc[32];
#pragma unroll
  for (int i = 0; i < 32; ++i) {
    const float id = (i == ln) ? 1.0f : 0.0f;
    Pc[i] = Mc[i]; Ec[i] = Mc[i] + id; Phc[i] = 0.5f * Mc[i] + id;
  }
  for (int k = 2; k <= 6; ++k) {
    const float rk = 1.0f / (float)k, rk1 = 1.0f / (float)(k + 1);
    float Np[32];
#pragma unroll
    for (int i = 0; i < 32; ++i) {
      float acc = 0.0f;
#pragma unroll
      for (int u = 0; u < 32; ++u) acc += Ep[i * 32 + u] * Pc[u];
      Np[i] = acc * rk;
    }
#pragma unroll
    for (int i = 0; i < 32; ++i) { Pc[i] = Np[i]; Ec[i] += Np[i]; Phc[i] += Np[i] * rk1; }
  }
  // squaring: E<-E*E ; Phi<-0.5*(E+I)*Phi  (Phi first, then E, to cap live registers)
  for (int q = 0; q < sq; ++q) {
#pragma unroll
    for (int i = 0; i < 32; ++i) Ep[i * 32 + ln] = Ec[i];
    __threadfence_block();
    {
      float Nph[32];
#pragma unroll
      for (int i = 0; i < 32; ++i) {
        float ap = 0.0f;
#pragma unroll
        for (int u = 0; u < 32; ++u) ap += Ep[i * 32 + u] * Phc[u];
        Nph[i] = 0.5f * (Phc[i] + ap);
      }
#pragma unroll
      for (int i = 0; i < 32; ++i) Phc[i] = Nph[i];
    }
    {
      float Ne[32];
#pragma unroll
      for (int i = 0; i < 32; ++i) {
        float ae = 0.0f;
#pragma unroll
        for (int u = 0; u < 32; ++u) ae += Ep[i * 32 + u] * Ec[u];
        Ne[i] = ae;
      }
#pragma unroll
      for (int i = 0; i < 32; ++i) Ec[i] = Ne[i];
    }
    __threadfence_block();
  }
#pragma unroll
  for (int i = 0; i < 32; ++i) Ep[i * 32 + ln] = Ec[i];   // final A_bar (unscaled)
  __threadfence_block();

  // smax via power iteration on A^T A
  float v = 0.17677669529663688f;
  for (int it = 0; it < 8; ++it) {
    float w = 0.0f;
#pragma unroll
    for (int u = 0; u < 32; ++u) w += Ep[ln * 32 + u] * __shfl(v, u, 32);
    float ua = 0.0f;
#pragma unroll
    for (int u = 0; u < 32; ++u) ua += Ep[u * 32 + ln] * __shfl(w, u, 32);
    float nn = ua * ua;
#pragma unroll
    for (int o = 16; o > 0; o >>= 1) nn += __shfl_xor(nn, o, 32);
    v = ua * rsqrtf(fmaxf(nn, 1e-30f));
  }
  float w = 0.0f;
#pragma unroll
  for (int u = 0; u < 32; ++u) w += Ep[ln * 32 + u] * __shfl(v, u, 32);
  float s2 = w * w;
#pragma unroll
  for (int o = 16; o > 0; o >>= 1) s2 += __shfl_xor(s2, o, 32);
  const float smax = sqrtf(s2);
  const float sc = fminf(0.999f / (smax + 1e-12f), 1.0f);

  // phi_qv_i = sum_j Phi[i][j]*qv_j  (cross-lane reduction per row)
  float pq = 0.0f;
#pragma unroll
  for (int i = 0; i < 32; ++i) {
    float r = Phc[i] * qv;
#pragma unroll
    for (int o = 16; o > 0; o >>= 1) r += __shfl_xor(r, o, 32);
    if (ln == i) pq = r;
  }
  // s_lin = sc*A_bar@state + dlt*phi_qv
  float sl = dlt * pq;
#pragma unroll
  for (int u = 0; u < 32; ++u) sl += sc * Ep[ln * 32 + u] * stateL[b * DST + u];
  eib[b * 160 + ln] = sl;
  for (int j = ln; j < DMODEL; j += 32) eib[b * 160 + DST + j] = hbuf[b * DMODEL + j];
  __threadfence_block();

  // mu = gelu(ei@em1)@em2 + b
  float hid[4];
#pragma unroll
  for (int u = 0; u < 4; ++u) {
    const int c = ln + 32 * u;
    float acc = em1_b[c];
#pragma unroll 8
    for (int k = 0; k < 160; ++k) acc += eib[b * 160 + k] * em1_w[k * DMODEL + c];
    hid[u] = gelu_f(acc);
  }
#pragma unroll
  for (int u = 0; u < 4; ++u) hidb[b * DMODEL + ln + 32 * u] = hid[u];
  __threadfence_block();
  float muv = em2_b[ln];
#pragma unroll 8
  for (int k = 0; k < DMODEL; ++k) muv += hidb[b * DMODEL + k] * em2_w[k * DST + ln];
  __threadfence_block();

  // logvar = clip(gelu(ei@el1)@el2 + b, -8, 8)
#pragma unroll
  for (int u = 0; u < 4; ++u) {
    const int c = ln + 32 * u;
    float acc = el1_b[c];
#pragma unroll 8
    for (int k = 0; k < 160; ++k) acc += eib[b * 160 + k] * el1_w[k * DMODEL + c];
    hid[u] = gelu_f(acc);
  }
#pragma unroll
  for (int u = 0; u < 4; ++u) hidb[b * DMODEL + ln + 32 * u] = hid[u];
  __threadfence_block();
  float lv = el2_b[ln];
#pragma unroll 8
  for (int k = 0; k < DMODEL; ++k) lv += hidb[b * DMODEL + k] * el2_w[k * DST + ln];
  lv = fminf(fmaxf(lv, -8.0f), 8.0f);

  float pm = pp_b[ln];
#pragma unroll
  for (int k = 0; k < DST; ++k) pm += stateL[b * DST + k] * pp_w[k * DST + ln];
  const float plv = p_lv[ln];
  const float dm = muv - pm;
  float kl = 0.5f * (plv - lv + (__expf(lv) + dm * dm) * __expf(-plv) - 1.0f);
#pragma unroll
  for (int o = 16; o > 0; o >>= 1) kl += __shfl_xor(kl, o, 32);
  kl *= (1.0f / 32.0f);

  stateL[b * DST + ln] = muv;
  if (ln == 0) rateL[b] = kl;
  __threadfence_block();
}

// ---------------- Kernel 2: persistent single-WGP sequential scan --------------------------
struct ScanP {
  const float *h, *g1_b, *g2_b, *g3_b, *A_base, *dt;
  const float *em1_w, *em1_b, *em2_w, *em2_b;
  const float *el1_w, *el1_b, *el2_w, *el2_b;
  const float *pp_w, *pp_b, *p_lv;
  const _Float16 *wG1, *wG2, *wG3;
  float *Bm, *state_out;
};

__global__ __launch_bounds__(512) void mps_scan(ScanP p) {
  __shared__ float    hbuf[NB * DMODEL];      // h_t f32 (TDM target)
  __shared__ _Float16 hf[NB * DMODEL];        // h_t f16
  __shared__ _Float16 act1[NB * DMODEL];
  __shared__ _Float16 act2[NB * DMODEL];
  __shared__ float    binsL[NB * NBINS];
  __shared__ float    Eb[NB * 32 * 32];       // per-batch 32x32 shared matrix (Ms, then e^M)
  __shared__ float    stateL[NB * DST];
  __shared__ float    rateL[NB];
  __shared__ float    eib[NB * (DST + DMODEL)];
  __shared__ float    hidb[NB * DMODEL];
  __shared__ float    g1bL[DMODEL];
  __shared__ float    g2bL[DMODEL];
  __shared__ float    g3bL[G3N];              // bias resident in LDS: no global loads in hot loop

  const int tid = threadIdx.x, wv = tid >> 5, ln = tid & 31;
  const int khi = ln >> 4;

  for (int i = tid; i < NB * DST; i += 512) stateL[i] = 0.0f;
  if (tid < NB) rateL[tid] = 0.0f;
  if (tid < DMODEL) { g1bL[tid] = p.g1_b[tid]; g2bL[tid] = p.g2_b[tid]; }
  for (int i = tid; i < G3N; i += 512) g3bL[i] = p.g3_b[i];
  __syncthreads();

  for (int t = 0; t < LSEQ; ++t) {
    // ---- stage h_t (16x128 f32) into LDS via the Tensor Data Mover ----
#if HAVE_TDM
    if (wv == 0) {
      const unsigned long long ga = (unsigned long long)(const void*)(p.h + (size_t)t * DMODEL);
      const unsigned int lds = (unsigned int)(unsigned long long)(&hbuf[0]);
      u32x4 g0 = {0u, 0u, 0u, 0u};
      g0[0] = 1u;                                            // count=1, user descriptor
      g0[1] = lds;                                           // lds_addr
      g0[2] = (unsigned int)(ga & 0xFFFFFFFFull);            // global_addr[95:64]
      g0[3] = (unsigned int)((ga >> 32) & 0x1FFFFFFull) | (2u << 30);  // addr hi | type=2
      i32x8 g1 = {0, 0, 0, 0, 0, 0, 0, 0};
      g1[0] = (int)(2u << 16);                               // data_size = 4B
      g1[1] = (int)(128u << 16);                             // tensor_dim0 = 128 (lo16 @bit48)
      g1[2] = (int)(16u << 16);                              // tensor_dim1 = 16  (lo16 @bit80)
      g1[3] = (int)(128u << 16);                             // tile_dim0 = 128
      g1[4] = (int)16;                                       // tile_dim1 = 16, tile_dim2 = 0
      g1[5] = (int)(LSEQ * DMODEL);                          // tensor_dim0_stride = 12288
      i32x4 z4 = {0, 0, 0, 0};
      i32x8 z8 = {0, 0, 0, 0, 0, 0, 0, 0};
      __builtin_amdgcn_tensor_load_to_lds(g0, g1, z4, z4, z8, 0);
      __builtin_amdgcn_s_wait_tensorcnt(0);
    }
#else
    for (int i = tid; i < NB * DMODEL; i += 512)
      hbuf[i] = p.h[((size_t)(i >> 7) * LSEQ + t) * DMODEL + (i & 127)];
#endif
    __syncthreads();
    for (int i = tid; i < NB * DMODEL; i += 512) hf[i] = (_Float16)hbuf[i];
    __syncthreads();

    // ---- G1: (16x128) @ g1_w(128x128), gelu -> act1 ----
    if (wv < 8) {
      v16h a[4];
#pragma unroll
      for (int kc = 0; kc < 4; ++kc) load_afrag(hf, kc, ln, a[kc]);
      v8f c = wmma_kx128(a, p.wG1, wv, ln);
      const int col = wv * 16 + (ln & 15);
      const float bv = g1bL[col];
#pragma unroll
      for (int r = 0; r < 8; ++r) act1[(r + khi * 8) * DMODEL + col] = (_Float16)gelu_f(c[r] + bv);
    }
    __syncthreads();

    // ---- G2 ----
    if (wv < 8) {
      v16h a[4];
#pragma unroll
      for (int kc = 0; kc < 4; ++kc) load_afrag(act1, kc, ln, a[kc]);
      v8f c = wmma_kx128(a, p.wG2, wv, ln);
      const int col = wv * 16 + (ln & 15);
      const float bv = g2bL[col];
#pragma unroll
      for (int r = 0; r < 8; ++r) act2[(r + khi * 8) * DMODEL + col] = (_Float16)gelu_f(c[r] + bv);
    }
    __syncthreads();

    // ---- G3: (16x128) @ g3_w[:, :4104]; bins -> LDS, Bm=tanh(.)*0.5/(1+0.2*rate) -> ws ----
    {
      const float rr0 = rateL[0 + khi * 8];   // not used; keep rate reads simple below
      (void)rr0;
      v16h a3[4];
#pragma unroll
      for (int kc = 0; kc < 4; ++kc) load_afrag(act2, kc, ln, a3[kc]);
      float* __restrict__ bmOut = p.Bm;
      const _Float16* __restrict__ packG3 = p.wG3;
      for (int tile = wv; tile < G3T; tile += 16) {
        if (tile + 16 < G3T)
          __builtin_prefetch((const void*)(packG3 + ((size_t)(tile + 16) * 4 * 32) * 16), 0, 1);
        v8f c = wmma_kx128(a3, packG3, tile, ln);
        const int col = tile * 16 + (ln & 15);
        if (col < G3N) {
          const float bv = g3bL[col];
#pragma unroll
          for (int r = 0; r < 8; ++r) {
            const int row = r + khi * 8;
            const float v = c[r] + bv;
            if (col < NBINS) binsL[row * NBINS + col] = v;
            else bmOut[(size_t)row * 4096 + (col - NBINS)] =
                   tanhf(v) * 0.5f / (1.0f + 0.2f * rateL[row]);
          }
        }
      }
    }
    __threadfence();
    __syncthreads();

    // ---- per-batch recurrence: wave b == batch b (noinline; no block barriers inside) ----
    mps_step_batch(p.A_base, p.dt, p.em1_w, p.em1_b, p.em2_w, p.em2_b,
                   p.el1_w, p.el1_b, p.el2_w, p.el2_b, p.pp_w, p.pp_b, p.p_lv,
                   p.Bm, binsL, rateL, stateL, hbuf, &Eb[wv * 1024], eib, hidb, wv, ln);
    __syncthreads();
  }

  for (int i = tid; i < NB * DST; i += 512) p.state_out[i] = stateL[i];
}

// ---------------- Kernel 3: out = gelu(state@ph1+b1)@ph2 + b2 ------------------------------
__global__ __launch_bounds__(128) void mps_head(
    const float* __restrict__ state, const float* __restrict__ ph1_w,
    const float* __restrict__ ph1_b, const _Float16* __restrict__ wPh2,
    const float* __restrict__ ph2_b, float* __restrict__ out) {
  __shared__ float st[NB * DST];
  __shared__ _Float16 hidf[NB * DMODEL];
  const int tid = threadIdx.x, wv = tid >> 5, ln = tid & 31;
  for (int i = tid; i < NB * DST; i += 128) st[i] = state[i];
  __syncthreads();
  for (int i = tid; i < NB * DMODEL; i += 128) {
    const int b = i >> 7, c = i & 127;
    float acc = ph1_b[c];
#pragma unroll
    for (int k = 0; k < DST; ++k) acc += st[b * DST + k] * ph1_w[k * DMODEL + c];
    hidf[i] = (_Float16)gelu_f(acc);
  }
  __syncthreads();
  const int tile = blockIdx.x * 4 + wv;
  v16h a[4];
#pragma unroll
  for (int kc = 0; kc < 4; ++kc) load_afrag(hidf, kc, ln, a[kc]);
  v8f c = wmma_kx128(a, wPh2, tile, ln);
  const int col = tile * 16 + (ln & 15), khi = ln >> 4;
  const float bb = ph2_b[col];
#pragma unroll
  for (int r = 0; r < 8; ++r) out[(size_t)(r + khi * 8) * PH2N + col] = c[r] + bb;
}

// ---------------- host side ----------------------------------------------------------------
extern "C" void kernel_launch(void* const* d_in, const int* in_sizes, int n_in,
                              void* d_out, int out_size, void* d_ws, size_t ws_size,
                              hipStream_t stream) {
  (void)in_sizes; (void)n_in; (void)out_size; (void)ws_size;
  const float* x        = (const float*)d_in[0];
  const float* We       = (const float*)d_in[1];
  const float* be       = (const float*)d_in[2];
  const float* Wr       = (const float*)d_in[3];
  const float* br       = (const float*)d_in[4];
  const float* ln_g     = (const float*)d_in[5];
  const float* ln_b     = (const float*)d_in[6];
  const float* conv_w   = (const float*)d_in[7];
  const float* conv_b   = (const float*)d_in[8];
  const float* g1_w     = (const float*)d_in[9];
  const float* g1_b     = (const float*)d_in[10];
  const float* g2_w     = (const float*)d_in[11];
  const float* g2_b     = (const float*)d_in[12];
  const float* g3_w     = (const float*)d_in[13];
  const float* g3_b     = (const float*)d_in[14];
  const float* em1_w    = (const float*)d_in[15];
  const float* em1_b    = (const float*)d_in[16];
  const float* em2_w    = (const float*)d_in[17];
  const float* em2_b    = (const float*)d_in[18];
  const float* el1_w    = (const float*)d_in[19];
  const float* el1_b    = (const float*)d_in[20];
  const float* el2_w    = (const float*)d_in[21];
  const float* el2_b    = (const float*)d_in[22];
  const float* pp_w     = (const float*)d_in[23];
  const float* pp_b     = (const float*)d_in[24];
  const float* p_lv     = (const float*)d_in[25];
  const float* ph1_w    = (const float*)d_in[26];
  const float* ph1_b    = (const float*)d_in[27];
  const float* ph2_w    = (const float*)d_in[28];
  const float* ph2_b    = (const float*)d_in[29];
  const float* A_base   = (const float*)d_in[30];
  const float* dtab     = (const float*)d_in[31];

  char* w = (char*)d_ws;
  float*    ws_h     = (float*)(w);                 // 16*96*128*4   = 786432
  float*    ws_Bm    = (float*)(w + 786432);        // 16*4096*4    = 262144
  float*    ws_state = (float*)(w + 1048576);       // 16*32*4      = 2048
  _Float16* wG1      = (_Float16*)(w + 1050624);    // 8*4*32*16*2  = 32768
  _Float16* wG2      = (_Float16*)(w + 1083392);    // 32768
  _Float16* wG3      = (_Float16*)(w + 1116160);    // 257*4*32*16*2 = 1052672
  _Float16* wPh2     = (_Float16*)(w + 2168832);    // 252*4*32*16*2 = 1032192

  mps_prep_h<<<NB * LSEQ, 128, 0, stream>>>(x, We, be, Wr, br, ln_g, ln_b, conv_w, conv_b, ws_h);
  mps_pack<<<16 + G3T + PH2T, 128, 0, stream>>>(g1_w, g2_w, g3_w, ph2_w, wG1, wG2, wG3, wPh2);

  ScanP P;
  P.h = ws_h; P.g1_b = g1_b; P.g2_b = g2_b; P.g3_b = g3_b; P.A_base = A_base; P.dt = dtab;
  P.em1_w = em1_w; P.em1_b = em1_b; P.em2_w = em2_w; P.em2_b = em2_b;
  P.el1_w = el1_w; P.el1_b = el1_b; P.el2_w = el2_w; P.el2_b = el2_b;
  P.pp_w = pp_w; P.pp_b = pp_b; P.p_lv = p_lv;
  P.wG1 = wG1; P.wG2 = wG2; P.wG3 = wG3;
  P.Bm = ws_Bm; P.state_out = ws_state;
  mps_scan<<<1, 512, 0, stream>>>(P);

  mps_head<<<PH2T / 4, 128, 0, stream>>>(ws_state, ph1_w, ph1_b, wPh2, ph2_b, (float*)d_out);
}
